// mLSTMBlock_70428873719993
// MI455X (gfx1250) — compile-verified, matrix operand order checked by reading, not run on hardware
//
#include <hip/hip_runtime.h>
#include <cstdint>
#include <cstddef>

// ---------------------------------------------------------------------------
// mLSTM block forward for MI455X (gfx1250), bf16 WMMA GEMMs + f32 scan.
// B=4 T=512 D=1024 H=8 Dh=64 HD=512 P=2048 K=4
// Weights are cast+TRANSPOSED to bf16 once so that both WMMA operands are
// K-contiguous -> all LDS fragment reads are ds_load_b128 (no scalar gathers).
// GEMM tiles are DOUBLE-BUFFERED and staged with CDNA5 async global->LDS DMA
// (ASYNCcnt): DMA for tile i+1 overlaps WMMA compute on tile i.
// ---------------------------------------------------------------------------

#define NTOK 2048      // B*T
#define DIM  1024      // D
#define PDIM 2048      // P
#define HDIM 512       // HD
#define NH   8         // H
#define DH   64        // Dh

typedef __attribute__((ext_vector_type(16))) __bf16 v16bf;
typedef __attribute__((ext_vector_type(8)))  float  v8f;

union FragV {
    uint4 u[2];
    v16bf v;
};

// async global->LDS 128-bit DMA (per-lane), tracked by ASYNCcnt
__device__ __forceinline__ void async_copy_b128(unsigned lds_addr, const void* gaddr)
{
    asm volatile("global_load_async_to_lds_b128 %0, %1, off"
                 :: "v"(lds_addr), "v"(gaddr) : "memory");
}
__device__ __forceinline__ void wait_asynccnt0()
{
    asm volatile("s_wait_asynccnt 0x0" ::: "memory");
}

// ---------------------------------------------------------------------------
// cast f32 -> bf16 with transpose: in (R x C row-major) -> out (C x R)
// 32x32 tiles through LDS, coalesced on both sides. R, C multiples of 32.
// ---------------------------------------------------------------------------
__global__ __launch_bounds__(256) void cast_transpose_kernel(
    const float* __restrict__ in, __bf16* __restrict__ out, int R, int C)
{
    __shared__ __bf16 tile[32][33];
    const int c0 = blockIdx.x * 32;
    const int r0 = blockIdx.y * 32;
    const int tx = threadIdx.x & 31;
    const int ty = threadIdx.x >> 5;       // 8 rows per pass
#pragma unroll
    for (int i = 0; i < 32; i += 8)
        tile[ty + i][tx] = (__bf16)in[(size_t)(r0 + ty + i) * C + c0 + tx];
    __syncthreads();
#pragma unroll
    for (int i = 0; i < 32; i += 8)
        out[(size_t)(c0 + ty + i) * R + r0 + tx] = tile[tx][ty + i];
}

// ---------------------------------------------------------------------------
// LayerNorm over D=1024, one block per token, writes bf16
// ---------------------------------------------------------------------------
__global__ __launch_bounds__(256) void layernorm_kernel(
    const float* __restrict__ x, const float* __restrict__ w,
    const float* __restrict__ b, __bf16* __restrict__ xn)
{
    const int tok = blockIdx.x;
    const int tid = threadIdx.x;
    const float* xr = x + (size_t)tok * DIM;

    float s = 0.f, ss = 0.f;
    float v[4];
#pragma unroll
    for (int i = 0; i < 4; ++i) {
        v[i] = xr[tid + i * 256];
        s += v[i];
        ss += v[i] * v[i];
    }
    __shared__ float r1[256], r2[256];
    r1[tid] = s; r2[tid] = ss;
    __syncthreads();
    for (int st = 128; st > 0; st >>= 1) {
        if (tid < st) { r1[tid] += r1[tid + st]; r2[tid] += r2[tid + st]; }
        __syncthreads();
    }
    const float mu  = r1[0] * (1.0f / DIM);
    const float var = r2[0] * (1.0f / DIM) - mu * mu;
    const float inv = rsqrtf(var + 1e-6f);
#pragma unroll
    for (int i = 0; i < 4; ++i) {
        int idx = tid + i * 256;
        float y = (v[i] - mu) * inv * w[idx] + b[idx];
        xn[(size_t)tok * DIM + idx] = (__bf16)y;
    }
}

// ---------------------------------------------------------------------------
// WMMA GEMM: C(MxN,f32) = epilogue(A_bf16(MxK,row) @ Bt_bf16(NxK,row)^T + bias)
//  EPI 0: val = (acc+bias)*scale
//  EPI 1: val = sigmoid(acc+bias)
//  EPI 2: val = acc+bias+resid[idx]
// Workgroup tile 128x128, 8 waves (4x2) of 32x64; K-step 32.
// Double-buffered LDS tiles staged with global_load_async_to_lds_b128:
// DMA of tile i+1 overlaps the WMMAs of tile i; fragments via ds_load_b128.
// M,N multiples of 128; K multiple of 32.
// ---------------------------------------------------------------------------
#define TILE_K  32
#define T_STRIDE 40   // 32 K + 8 pad, keeps 16B alignment for c in {0,8,16,24}
#define BUF_ELEMS (128 * T_STRIDE)
#define BUF_BYTES (BUF_ELEMS * 2)

template <int EPI>
__global__ __launch_bounds__(256) void gemm_bf16_wmma(
    const __bf16* __restrict__ A, const __bf16* __restrict__ Bt,
    const float* __restrict__ bias, const float* __restrict__ resid,
    float* __restrict__ C, int M, int N, int K, float scale)
{
    __shared__ __bf16 As[2][BUF_ELEMS];
    __shared__ __bf16 Bs[2][BUF_ELEMS];

    const int tid  = threadIdx.x;
    const int lane = tid & 31;
    const int wave = tid >> 5;
    const int wm   = wave & 3;   // wave row (0..3) -> 32 rows each
    const int wn   = wave >> 2;  // wave col (0..1) -> 64 cols each
    const int row0 = blockIdx.y * 128;
    const int col0 = blockIdx.x * 128;

    const int lhalf = lane >> 4;   // 0 | 1
    const int lmod  = lane & 15;
    const int kof   = lhalf * 8;   // K offset within 16-lane half

    // per-thread staging coordinates (8 bf16 = 16B per transfer, x2 rows)
    const int lin0 = tid * 8;
    const int lin1 = (tid + 256) * 8;
    const int sr0 = lin0 >> 5, sc0 = lin0 & 31;
    const int sr1 = lin1 >> 5, sc1 = lin1 & 31;
    const unsigned ldsA0 = (unsigned)(uintptr_t)&As[0][sr0 * T_STRIDE + sc0];
    const unsigned ldsA1 = (unsigned)(uintptr_t)&As[0][sr1 * T_STRIDE + sc1];
    const unsigned ldsB0 = (unsigned)(uintptr_t)&Bs[0][sr0 * T_STRIDE + sc0];
    const unsigned ldsB1 = (unsigned)(uintptr_t)&Bs[0][sr1 * T_STRIDE + sc1];

    const __bf16* gA0 = A  + (size_t)(row0 + sr0) * K + sc0;
    const __bf16* gA1 = A  + (size_t)(row0 + sr1) * K + sc1;
    const __bf16* gB0 = Bt + (size_t)(col0 + sr0) * K + sc0;
    const __bf16* gB1 = Bt + (size_t)(col0 + sr1) * K + sc1;

    auto stage = [&](int k0, int buf) {
        const unsigned off = (unsigned)buf * BUF_BYTES;
        async_copy_b128(ldsA0 + off, gA0 + k0);
        async_copy_b128(ldsA1 + off, gA1 + k0);
        async_copy_b128(ldsB0 + off, gB0 + k0);
        async_copy_b128(ldsB1 + off, gB1 + k0);
    };

    v8f acc[2][4];
#pragma unroll
    for (int i = 0; i < 2; ++i)
#pragma unroll
        for (int j = 0; j < 4; ++j) {
            v8f z = {};
            acc[i][j] = z;
        }

    // preload tile 0
    stage(0, 0);
    wait_asynccnt0();
    __syncthreads();

    int cur = 0;
    for (int k0 = 0; k0 < K; k0 += TILE_K) {
        // issue DMA for the NEXT tile into the other buffer (overlaps compute)
        if (k0 + TILE_K < K) {
            stage(k0 + TILE_K, cur ^ 1);
            // L2 prefetch two tiles ahead (global_prefetch_b8 path)
            __builtin_prefetch(gA0 + k0 + 2 * TILE_K, 0, 1);
            __builtin_prefetch(gB0 + k0 + 2 * TILE_K, 0, 1);
        }

        // --- fragments from current buffer (K-contiguous for A and B) ---
        FragV af[2], bfr[4];
#pragma unroll
        for (int mt = 0; mt < 2; ++mt) {
            const int r = wm * 32 + mt * 16 + lmod;
            af[mt].u[0] = *reinterpret_cast<const uint4*>(&As[cur][r * T_STRIDE + kof]);
            af[mt].u[1] = *reinterpret_cast<const uint4*>(&As[cur][r * T_STRIDE + 16 + kof]);
        }
#pragma unroll
        for (int nt = 0; nt < 4; ++nt) {
            const int r = wn * 64 + nt * 16 + lmod;
            bfr[nt].u[0] = *reinterpret_cast<const uint4*>(&Bs[cur][r * T_STRIDE + kof]);
            bfr[nt].u[1] = *reinterpret_cast<const uint4*>(&Bs[cur][r * T_STRIDE + 16 + kof]);
        }
        // --- 8 WMMAs per wave per K-step (DMA for next tile in flight) ---
#pragma unroll
        for (int mt = 0; mt < 2; ++mt)
#pragma unroll
            for (int nt = 0; nt < 4; ++nt)
                acc[mt][nt] = __builtin_amdgcn_wmma_f32_16x16x32_bf16(
                    false, af[mt].v, false, bfr[nt].v,
                    (short)0, acc[mt][nt], false, false);

        // next tile fully in LDS (own wave) + all waves done with both ops
        wait_asynccnt0();
        __syncthreads();
        cur ^= 1;
    }

    // --- epilogue + store (C layout: lane<16 -> M=vgpr, lane>=16 -> M=8+vgpr) ---
#pragma unroll
    for (int mt = 0; mt < 2; ++mt) {
#pragma unroll
        for (int nt = 0; nt < 4; ++nt) {
            const int col = col0 + wn * 64 + nt * 16 + lmod;
            const float bv = bias[col];
#pragma unroll
            for (int vv = 0; vv < 8; ++vv) {
                const int row = row0 + wm * 32 + mt * 16 + lhalf * 8 + vv;
                const size_t idx = (size_t)row * N + col;
                float val = acc[mt][nt][vv] + bv;
                if (EPI == 0) {
                    val *= scale;
                } else if (EPI == 1) {
                    val = 1.0f / (1.0f + __expf(-val));
                } else {
                    val += resid[idx];
                }
                C[idx] = val;
            }
        }
    }
}

// ---------------------------------------------------------------------------
// Causal conv (K=4) along the feature axis + SiLU; also emits xt as bf16.
// ---------------------------------------------------------------------------
__global__ __launch_bounds__(256) void conv_silu_kernel(
    const float* __restrict__ xt, const float* __restrict__ cw,
    const float* __restrict__ cb, __bf16* __restrict__ xc_bf,
    __bf16* __restrict__ xt_bf)
{
    const int tok = blockIdx.y;
    const int p   = blockIdx.x * 256 + threadIdx.x;
    const size_t base = (size_t)tok * PDIM;
    const float w0 = cw[0], w1 = cw[1], w2 = cw[2], w3 = cw[3];

    const float x3 = xt[base + p];
    float acc = cb[0] + w3 * x3;
    if (p >= 1) acc += w2 * xt[base + p - 1];
    if (p >= 2) acc += w1 * xt[base + p - 2];
    if (p >= 3) acc += w0 * xt[base + p - 3];
    const float sv = acc / (1.0f + __expf(-acc));   // silu
    xc_bf[base + p] = (__bf16)sv;
    xt_bf[base + p] = (__bf16)x3;
}

// ---------------------------------------------------------------------------
// i/f gate projections (P x 8 each) + softcap(15) — tiny, plain VALU.
// ---------------------------------------------------------------------------
__global__ __launch_bounds__(256) void gates_kernel(
    const __bf16* __restrict__ xc, const float* __restrict__ Wi,
    const float* __restrict__ bi, const float* __restrict__ Wf,
    const float* __restrict__ bfv, float* __restrict__ ipre,
    float* __restrict__ fpre)
{
    const int tok = blockIdx.x;
    const int tid = threadIdx.x;
    float pi[NH] = {}, pf[NH] = {};
    for (int p = tid; p < PDIM; p += 256) {
        const float xv = (float)xc[(size_t)tok * PDIM + p];
#pragma unroll
        for (int g = 0; g < NH; ++g) {
            pi[g] += xv * Wi[p * NH + g];
            pf[g] += xv * Wf[p * NH + g];
        }
    }
    __shared__ float red[256];
    for (int g = 0; g < NH; ++g) {
        red[tid] = pi[g];
        __syncthreads();
        for (int s = 128; s > 0; s >>= 1) {
            if (tid < s) red[tid] += red[tid + s];
            __syncthreads();
        }
        if (tid == 0) {
            float v = red[0] + bi[g];
            ipre[(size_t)tok * NH + g] = 15.0f * tanhf(v * (1.0f / 15.0f));
        }
        __syncthreads();
        red[tid] = pf[g];
        __syncthreads();
        for (int s = 128; s > 0; s >>= 1) {
            if (tid < s) red[tid] += red[tid + s];
            __syncthreads();
        }
        if (tid == 0) {
            float v = red[0] + bfv[g];
            fpre[(size_t)tok * NH + g] = 15.0f * tanhf(v * (1.0f / 15.0f));
        }
        __syncthreads();
    }
}

// ---------------------------------------------------------------------------
// Sequential mLSTM scan. One block per (b,h); C(64x64) state in registers:
// thread t owns row d=t/4, e-range [(t%4)*16, +16). n held by threads < 64.
// ---------------------------------------------------------------------------
__global__ __launch_bounds__(256) void scan_kernel(
    const float* __restrict__ q, const float* __restrict__ k,
    const float* __restrict__ v, const float* __restrict__ ipre,
    const float* __restrict__ fpre, float* __restrict__ hseq)
{
    const int bh = blockIdx.x;           // 0..31
    const int b  = bh >> 3;
    const int h  = bh & 7;
    const int tid = threadIdx.x;
    const int d  = tid >> 2;
    const int e0 = (tid & 3) * 16;

    float c[16];
#pragma unroll
    for (int j = 0; j < 16; ++j) c[j] = 0.0f;
    float ne = 1.0f;                     // n init = ones
    float m  = 0.0f;

    __shared__ float qs[64], ks[64], vs[64];
    __shared__ float part[64][5];
    __shared__ float red[64];
    __shared__ float gi, gf, den;

    for (int t = 0; t < 512; ++t) {
        const size_t gidx = (size_t)(b * 512 + t) * NH + h;
        const size_t base = gidx * DH;
        if (tid < 64) {
            qs[tid] = q[base + tid];
            ks[tid] = k[base + tid];
            vs[tid] = v[base + tid];
        }
        if (tid == 0) { gi = ipre[gidx]; gf = fpre[gidx]; }
        __syncthreads();

        const float it = gi, ft = gf;
        const float mnew = fmaxf(ft + m, it);
        const float ie = __expf(it - mnew);
        const float fe = __expf(ft - mnew + m);
        m = mnew;

        const float vd = vs[d];
        float ps = 0.0f;
#pragma unroll
        for (int j = 0; j < 16; ++j) {
            const int e = e0 + j;
            c[j] = fe * c[j] + ie * vd * ks[e];
            ps += c[j] * qs[e];
        }
        part[d][tid & 3] = ps;
        if (tid < 64) {
            ne = fe * ne + ie * ks[tid];
            red[tid] = ne * qs[tid];
        }
        __syncthreads();
        if (tid == 0) {
            float s = 0.0f;
            for (int e = 0; e < 64; ++e) s += red[e];
            den = fmaxf(s, 1.0f);
        }
        __syncthreads();
        if (tid < 64) {
            const float num = part[tid][0] + part[tid][1] + part[tid][2] + part[tid][3];
            hseq[base + tid] = num / den;
        }
        __syncthreads();
    }
}

// ---------------------------------------------------------------------------
// h = o * hseq; per-head LayerNorm over Dh; z = (hn*hw+hb + skip) * silu(r)
// One block (512 threads) per token; thread = (h,d).
// ---------------------------------------------------------------------------
__global__ __launch_bounds__(512) void combine_kernel(
    const float* __restrict__ o, const float* __restrict__ hseq,
    const float* __restrict__ skip, const float* __restrict__ r,
    const float* __restrict__ hlnw, const float* __restrict__ hlnb,
    __bf16* __restrict__ z)
{
    const int tok = blockIdx.x;
    const int tid = threadIdx.x;          // 0..511
    const int d   = tid & 63;
    const size_t idx = (size_t)tok * HDIM + tid;

    const float hv = o[idx] * hseq[idx];
    __shared__ float s1[512], s2[512];
    s1[tid] = hv; s2[tid] = hv * hv;
    __syncthreads();
    for (int st = 32; st > 0; st >>= 1) {
        if (d < st) { s1[tid] += s1[tid + st]; s2[tid] += s2[tid + st]; }
        __syncthreads();
    }
    const int hb0 = tid & ~63;
    const float mu  = s1[hb0] * (1.0f / DH);
    const float var = s2[hb0] * (1.0f / DH) - mu * mu;
    const float hn  = (hv - mu) * rsqrtf(var + 1e-6f) * hlnw[tid] + hlnb[tid];
    const float rv  = r[idx];
    const float sr  = rv / (1.0f + __expf(-rv));
    z[idx] = (__bf16)((hn + skip[idx]) * sr);
}

// ---------------------------------------------------------------------------
// host launcher
// ---------------------------------------------------------------------------
extern "C" void kernel_launch(void* const* d_in, const int* in_sizes, int n_in,
                              void* d_out, int out_size, void* d_ws, size_t ws_size,
                              hipStream_t stream)
{
    const float* x     = (const float*)d_in[0];
    const float* ln_w  = (const float*)d_in[1];
    const float* ln_b  = (const float*)d_in[2];
    const float* hln_w = (const float*)d_in[3];
    const float* hln_b = (const float*)d_in[4];
    const float* Wl    = (const float*)d_in[5];
    const float* bl    = (const float*)d_in[6];
    const float* Wr    = (const float*)d_in[7];
    const float* br    = (const float*)d_in[8];
    const float* conv_w= (const float*)d_in[9];
    const float* conv_b= (const float*)d_in[10];
    const float* Wskip = (const float*)d_in[11];
    const float* bskip = (const float*)d_in[12];
    const float* Wi    = (const float*)d_in[13];
    const float* bi    = (const float*)d_in[14];
    const float* Wf    = (const float*)d_in[15];
    const float* bfv   = (const float*)d_in[16];
    const float* Wo    = (const float*)d_in[17];
    const float* bo    = (const float*)d_in[18];
    const float* Wq    = (const float*)d_in[19];
    const float* bq    = (const float*)d_in[20];
    const float* Wk    = (const float*)d_in[21];
    const float* bk    = (const float*)d_in[22];
    const float* Wv    = (const float*)d_in[23];
    const float* bv    = (const float*)d_in[24];
    const float* Wd    = (const float*)d_in[25];
    const float* bd    = (const float*)d_in[26];

    // --- workspace carve (whole working set fits in MI455X's 192MB L2) ---
    char* ws = (char*)d_ws;
    auto carve = [&](size_t bytes) {
        char* p = ws;
        ws += (bytes + 255) & ~(size_t)255;
        return p;
    };
    // transposed bf16 weights (N x K layout)
    __bf16* WlT    = (__bf16*)carve((size_t)DIM * PDIM * 2);   // 2048 x 1024
    __bf16* WrT    = (__bf16*)carve((size_t)DIM * HDIM * 2);   // 512 x 1024
    __bf16* WskipT = (__bf16*)carve((size_t)PDIM * HDIM * 2);  // 512 x 2048
    __bf16* WqT    = (__bf16*)carve((size_t)PDIM * HDIM * 2);
    __bf16* WkT    = (__bf16*)carve((size_t)PDIM * HDIM * 2);
    __bf16* WvT    = (__bf16*)carve((size_t)PDIM * HDIM * 2);
    __bf16* WoT    = (__bf16*)carve((size_t)PDIM * HDIM * 2);
    __bf16* WdT    = (__bf16*)carve((size_t)HDIM * DIM * 2);   // 1024 x 512
    __bf16* xn_bf  = (__bf16*)carve((size_t)NTOK * DIM * 2);
    float*  xt     = (float*) carve((size_t)NTOK * PDIM * 4);
    __bf16* xt_bf  = (__bf16*)carve((size_t)NTOK * PDIM * 2);
    __bf16* xc_bf  = (__bf16*)carve((size_t)NTOK * PDIM * 2);
    float*  rbuf   = (float*) carve((size_t)NTOK * HDIM * 4);
    float*  skip   = (float*) carve((size_t)NTOK * HDIM * 4);
    float*  qb     = (float*) carve((size_t)NTOK * HDIM * 4);
    float*  kb     = (float*) carve((size_t)NTOK * HDIM * 4);
    float*  vb     = (float*) carve((size_t)NTOK * HDIM * 4);
    float*  ob     = (float*) carve((size_t)NTOK * HDIM * 4);
    float*  ipre   = (float*) carve((size_t)NTOK * NH * 4);
    float*  fpre   = (float*) carve((size_t)NTOK * NH * 4);
    float*  hseq   = (float*) carve((size_t)NTOK * HDIM * 4);
    __bf16* z_bf   = (__bf16*)carve((size_t)NTOK * HDIM * 2);
    (void)ws_size; (void)in_sizes; (void)n_in; (void)out_size;

    // cast + transpose weights: src is (K x N) row-major -> dst (N x K) bf16
    auto castT = [&](const float* src, __bf16* dst, int K, int N) {
        cast_transpose_kernel<<<dim3(N / 32, K / 32), 256, 0, stream>>>(src, dst, K, N);
    };
    castT(Wl,    WlT,    DIM,  PDIM);
    castT(Wr,    WrT,    DIM,  HDIM);
    castT(Wskip, WskipT, PDIM, HDIM);
    castT(Wq,    WqT,    PDIM, HDIM);
    castT(Wk,    WkT,    PDIM, HDIM);
    castT(Wv,    WvT,    PDIM, HDIM);
    castT(Wo,    WoT,    PDIM, HDIM);
    castT(Wd,    WdT,    HDIM, DIM);

    layernorm_kernel<<<NTOK, 256, 0, stream>>>(x, ln_w, ln_b, xn_bf);

    // xt = xn @ Wl + bl   (2048x1024x2048)
    gemm_bf16_wmma<0><<<dim3(PDIM / 128, NTOK / 128), 256, 0, stream>>>(
        xn_bf, WlT, bl, nullptr, xt, NTOK, PDIM, DIM, 1.0f);
    // r = xn @ Wr + br
    gemm_bf16_wmma<0><<<dim3(HDIM / 128, NTOK / 128), 256, 0, stream>>>(
        xn_bf, WrT, br, nullptr, rbuf, NTOK, HDIM, DIM, 1.0f);

    conv_silu_kernel<<<dim3(PDIM / 256, NTOK), 256, 0, stream>>>(
        xt, conv_w, conv_b, xc_bf, xt_bf);

    // skip / q / k (k scaled by 1/sqrt(Dh))
    gemm_bf16_wmma<0><<<dim3(HDIM / 128, NTOK / 128), 256, 0, stream>>>(
        xc_bf, WskipT, bskip, nullptr, skip, NTOK, HDIM, PDIM, 1.0f);
    gemm_bf16_wmma<0><<<dim3(HDIM / 128, NTOK / 128), 256, 0, stream>>>(
        xc_bf, WqT, bq, nullptr, qb, NTOK, HDIM, PDIM, 1.0f);
    gemm_bf16_wmma<0><<<dim3(HDIM / 128, NTOK / 128), 256, 0, stream>>>(
        xc_bf, WkT, bk, nullptr, kb, NTOK, HDIM, PDIM, 0.125f);
    // v = xt @ Wv, o = sigmoid(xt @ Wo)
    gemm_bf16_wmma<0><<<dim3(HDIM / 128, NTOK / 128), 256, 0, stream>>>(
        xt_bf, WvT, bv, nullptr, vb, NTOK, HDIM, PDIM, 1.0f);
    gemm_bf16_wmma<1><<<dim3(HDIM / 128, NTOK / 128), 256, 0, stream>>>(
        xt_bf, WoT, bo, nullptr, ob, NTOK, HDIM, PDIM, 1.0f);

    gates_kernel<<<NTOK, 256, 0, stream>>>(xc_bf, Wi, bi, Wf, bfv, ipre, fpre);

    scan_kernel<<<32, 256, 0, stream>>>(qb, kb, vb, ipre, fpre, hseq);

    combine_kernel<<<NTOK, 512, 0, stream>>>(ob, hseq, skip, rbuf, hln_w, hln_b, z_bf);

    // out = z @ Wd + bd + x
    gemm_bf16_wmma<2><<<dim3(DIM / 128, NTOK / 128), 256, 0, stream>>>(
        z_bf, WdT, bd, x, (float*)d_out, NTOK, DIM, HDIM, 1.0f);
}
